// FlowLayer_71554155151377
// MI455X (gfx1250) — matrix operand order repaired and checked by reading, not compiled, not run on hardware
//
#include <hip/hip_runtime.h>
#include <hip/hip_bf16.h>

// ---------------------------------------------------------------------------
// MI455X (gfx1250) implementation.
//  * All channel contractions (1x1 convs / pointwise convs) -> v_wmma_f32_16x16x32_f16
//  * Activations kept NHWC f32 so the 24-step bilinear-warp scan does fully
//    coalesced float4 gathers (one wave per pixel, 128 channels = 32 lanes x float4).
//  * Entire working set (~130 MB) is L2-resident on the 192 MB L2.
// ---------------------------------------------------------------------------

#define BN    2
#define CINc  256
#define COUTc 128
#define HHc   128
#define WWc   128
#define HWc   (HHc * WWc)     // 16384
#define NPIX  (BN * HWc)      // 32768
#define NITER 24              // 3*H/(4*PPI)

typedef __attribute__((ext_vector_type(8)))  _Float16 v8h;
typedef __attribute__((ext_vector_type(16))) _Float16 v16h;
typedef __attribute__((ext_vector_type(8)))  float    v8f;

union H16 { v16h v; v8h h[2]; };

__device__ __forceinline__ float sigmoidf(float x) {
    return 1.0f / (1.0f + __expf(-x));
}

// 16x16 output tile, D = A(16xK) * B(KxN=16) accumulated f32.
// A: row-major f16, leading dim lda.  Per ISA 7.12.2 (16-bit A 16x32):
//   lane l<16 : row=l,   VGPR0..3 hold K=kb..kb+7 (kb=0), VGPR4..7 K=16..23
//   lane l>=16: row=l-16, kb=8 -> K=8..15 and K=24..31
// B source is the weight matrix stored [cout][cin] row-major == B column-major:
//   lane l<16 : col=l,   K = k0..k0+7 and k0+16..k0+23   (two 16B loads)
//   lane l>=16: col=l-16,K = k0+8..k0+15 and k0+24..k0+31
__device__ __forceinline__ v8f wmma_kloop(const _Float16* __restrict__ A, int lda,
                                          const _Float16* __restrict__ Wt, int K,
                                          int lane, v8f c) {
    const int row = lane & 15;
    const int kb  = (lane >> 4) << 3;   // 0 or 8  (A half-K offset)
    const int kw  = (lane >> 4) << 3;   // 0 or 8  (B half-K offset)
    const _Float16* ar = A  + (size_t)row * lda + kb;
    const _Float16* wr = Wt + (size_t)row * K   + kw;  // col == lane&15 == row
    for (int k0 = 0; k0 < K; k0 += 32) {
        H16 a;
        a.h[0] = *(const v8h*)(ar + k0);
        a.h[1] = *(const v8h*)(ar + k0 + 16);
        H16 bb;
        bb.h[0] = *(const v8h*)(wr + k0);
        bb.h[1] = *(const v8h*)(wr + k0 + 16);
        v16h b = bb.v;
        c = __builtin_amdgcn_wmma_f32_16x16x32_f16(false, a.v, false, b,
                                                   (short)0, c, false, false);
    }
    return c;
}

// ---------------------------------------------------------------------------
__global__ void k_f32_to_f16(const float* __restrict__ in,
                             _Float16* __restrict__ out, int n) {
    int i = blockIdx.x * blockDim.x + threadIdx.x;
    if (i < n) out[i] = (_Float16)in[i];
}

// x [B,CIN,H,W] f32  ->  xT [B*HW, CIN] f16 (NHWC), 32x32 LDS tile transpose
__global__ void k_transpose_x(const float* __restrict__ x,
                              _Float16* __restrict__ xt) {
    __shared__ float tile[32][33];
    const int b   = blockIdx.z;
    const int hw0 = blockIdx.x * 32;
    const int c0  = blockIdx.y * 32;
    const int tx = threadIdx.x, ty = threadIdx.y;
#pragma unroll
    for (int k = 0; k < 4; ++k) {
        int c = c0 + ty + 8 * k;
        tile[ty + 8 * k][tx] = x[(size_t)(b * CINc + c) * HWc + hw0 + tx];
    }
    __syncthreads();
#pragma unroll
    for (int k = 0; k < 4; ++k) {
        int hw = hw0 + ty + 8 * k;
        xt[(size_t)(b * HWc + hw) * CINc + c0 + tx] = (_Float16)tile[tx][ty + 8 * k];
    }
}

// m[pix, cout] = sigmoid( sum_cin xT[pix,cin] * w[cout,cin] ),  f32 NHWC out
__global__ void k_conv1x1_sigmoid(const _Float16* __restrict__ xt,
                                  const _Float16* __restrict__ wh,
                                  float* __restrict__ m) {
    const int lane  = threadIdx.x & 31;
    const int flat  = blockIdx.x * 8 + (threadIdx.x >> 5);
    const int coutT = (flat & 7) << 4;
    const int pixT  = (flat >> 3) << 4;
    v8f c = {};
    c = wmma_kloop(xt + (size_t)pixT * CINc, CINc,
                   wh + (size_t)coutT * CINc, CINc, lane, c);
    const int colc = coutT + (lane & 15);
    const int prow = pixT + ((lane >> 4) << 3);   // C/D: M = r + 8*(lane>=16)
#pragma unroll
    for (int r = 0; r < 8; ++r)
        m[(size_t)(prow + r) * COUTc + colc] = sigmoidf(c[r]);
}

// Per-pixel constant bilinear corner indices/weights for grid1(+g) / grid2(-g)
__global__ void k_grid_setup(const float* __restrict__ grads,
                             int4* __restrict__ idx1, float4* __restrict__ wt1,
                             int4* __restrict__ idx2, float4* __restrict__ wt2) {
    int p = blockIdx.x * blockDim.x + threadIdx.x;
    if (p >= NPIX) return;
    const int b = p >> 14, hw = p & (HWc - 1);
    const int h = hw >> 7, w = hw & (WWc - 1);
    const float gx = grads[(size_t)(b * 2 + 0) * HWc + hw];
    const float gy = grads[(size_t)(b * 2 + 1) * HWc + hw];
    const float sx = 1.5f * (WWc - 1) / (2.0f * NITER);
    const float sy = 1.5f * (HHc - 1) / (2.0f * NITER);
    const int bOff = b * HWc;
    for (int s = 0; s < 2; ++s) {
        const float sgn = s ? -1.0f : 1.0f;
        const float xf = (float)w + sgn * sx * gx;
        const float yf = (float)h + sgn * sy * gy;
        const float x0 = floorf(xf), y0 = floorf(yf);
        const float wx1 = xf - x0, wx0 = 1.0f - wx1;
        const float wy1 = yf - y0, wy0 = 1.0f - wy1;
        const int xi0 = (int)x0, yi0 = (int)y0, xi1 = xi0 + 1, yi1 = yi0 + 1;
        int4 id; float4 wv;
        {   // corner helper (zeros padding, clamped gather index)
            int xs[4] = {xi0, xi1, xi0, xi1};
            int ys[4] = {yi0, yi0, yi1, yi1};
            float ws[4] = {wx0 * wy0, wx1 * wy0, wx0 * wy1, wx1 * wy1};
            int* idp = &id.x; float* wvp = &wv.x;
            for (int k = 0; k < 4; ++k) {
                bool valid = (xs[k] >= 0) && (xs[k] < WWc) &&
                             (ys[k] >= 0) && (ys[k] < HHc);
                int xc = min(max(xs[k], 0), WWc - 1);
                int yc = min(max(ys[k], 0), HHc - 1);
                idp[k] = bOff + yc * WWc + xc;
                wvp[k] = valid ? ws[k] : 0.0f;
            }
        }
        if (s == 0) { idx1[p] = id; wt1[p] = wv; }
        else        { idx2[p] = id; wt2[p] = wv; }
    }
}

// One warp-scan step for both streams. One wave per (pixel,stream);
// lane covers 4 channels (float4) -> each corner gather is a coalesced 512B row.
__global__ void k_warp_step(const float4* __restrict__ a1o, float4* __restrict__ a1n,
                            float4* __restrict__ s1,
                            const float4* __restrict__ a2o, float4* __restrict__ a2n,
                            float4* __restrict__ s2,
                            const int4* __restrict__ idx1, const float4* __restrict__ wt1,
                            const int4* __restrict__ idx2, const float4* __restrict__ wt2,
                            int first) {
    const int lane = threadIdx.x & 31;
    const int flat = blockIdx.x * 8 + (threadIdx.x >> 5);
    const int strm = flat & 1;
    const int p    = flat >> 1;
    const float4* ao = strm ? a2o : a1o;
    float4* an = strm ? a2n : a1n;
    float4* sv = strm ? s2  : s1;
    const int4   id = (strm ? idx2 : idx1)[p];
    const float4 wv = (strm ? wt2  : wt1 )[p];
    const float4 v0 = ao[(size_t)id.x * 32 + lane];
    const float4 v1 = ao[(size_t)id.y * 32 + lane];
    const float4 v2 = ao[(size_t)id.z * 32 + lane];
    const float4 v3 = ao[(size_t)id.w * 32 + lane];
    float4 acc;
    acc.x = wv.x * v0.x + wv.y * v1.x + wv.z * v2.x + wv.w * v3.x;
    acc.y = wv.x * v0.y + wv.y * v1.y + wv.z * v2.y + wv.w * v3.y;
    acc.z = wv.x * v0.z + wv.y * v1.z + wv.z * v2.z + wv.w * v3.z;
    acc.w = wv.x * v0.w + wv.y * v1.w + wv.z * v2.w + wv.w * v3.w;
    const size_t o = (size_t)p * 32 + lane;
    an[o] = acc;
    const float4 sp = first ? ao[o] : sv[o];   // s starts at m (the iter-0 a_old)
    float4 sn; sn.x = sp.x + acc.x; sn.y = sp.y + acc.y;
    sn.z = sp.z + acc.z; sn.w = sp.w + acc.w;
    sv[o] = sn;
}

// sigmoid + channel-concat + depthwise 3x3 (groups=256) -> f16 NHWC [p][256]
__global__ void k_dw_merge(const float* __restrict__ s1, const float* __restrict__ s2,
                           const float* __restrict__ dw, _Float16* __restrict__ dcat) {
    const int t = blockIdx.x * blockDim.x + threadIdx.x;   // NPIX*256
    const int c256 = t & 255;
    const int p = t >> 8;
    const int c = c256 & 127;
    const float* src = (c256 & 128) ? s2 : s1;
    const int b = p >> 14, hw = p & (HWc - 1);
    const int h = hw >> 7, w = hw & (WWc - 1);
    float acc = 0.0f;
#pragma unroll
    for (int ky = 0; ky < 3; ++ky) {
        const int yy = h + ky - 1;
        if (yy < 0 || yy >= HHc) continue;
#pragma unroll
        for (int kx = 0; kx < 3; ++kx) {
            const int xx = w + kx - 1;
            if (xx < 0 || xx >= WWc) continue;
            const float v = src[(size_t)(b * HWc + yy * WWc + xx) * COUTc + c];
            acc += dw[c256 * 9 + ky * 3 + kx] * sigmoidf(v);
        }
    }
    dcat[(size_t)p * 256 + c256] = (_Float16)acc;
}

// merge pointwise 256->128 + bias, f16 NHWC out
__global__ void k_gemm_merge(const _Float16* __restrict__ dcat,
                             const _Float16* __restrict__ pwh,
                             const float* __restrict__ bias,
                             _Float16* __restrict__ tout) {
    const int lane  = threadIdx.x & 31;
    const int flat  = blockIdx.x * 8 + (threadIdx.x >> 5);
    const int coutT = (flat & 7) << 4;
    const int pixT  = (flat >> 3) << 4;
    v8f c = {};
    c = wmma_kloop(dcat + (size_t)pixT * 256, 256,
                   pwh + (size_t)coutT * 256, 256, lane, c);
    const int colc = coutT + (lane & 15);
    const float bb = bias[colc];
    const int prow = pixT + ((lane >> 4) << 3);
#pragma unroll
    for (int r = 0; r < 8; ++r)
        tout[(size_t)(prow + r) * COUTc + colc] = (_Float16)(c[r] + bb);
}

// post depthwise 3x3 (groups=128), f16 NHWC in/out
__global__ void k_dw_post(const _Float16* __restrict__ tin,
                          const float* __restrict__ dw, _Float16* __restrict__ d2) {
    const int t = blockIdx.x * blockDim.x + threadIdx.x;   // NPIX*128
    const int c = t & 127;
    const int p = t >> 7;
    const int b = p >> 14, hw = p & (HWc - 1);
    const int h = hw >> 7, w = hw & (WWc - 1);
    float acc = 0.0f;
#pragma unroll
    for (int ky = 0; ky < 3; ++ky) {
        const int yy = h + ky - 1;
        if (yy < 0 || yy >= HHc) continue;
#pragma unroll
        for (int kx = 0; kx < 3; ++kx) {
            const int xx = w + kx - 1;
            if (xx < 0 || xx >= WWc) continue;
            acc += dw[c * 9 + ky * 3 + kx] *
                   (float)tin[(size_t)(b * HWc + yy * WWc + xx) * COUTc + c];
        }
    }
    d2[(size_t)p * COUTc + c] = (_Float16)acc;
}

// post pointwise 128->128 + bias, scatter to NCHW f32 d_out
__global__ void k_gemm_post(const _Float16* __restrict__ d2,
                            const _Float16* __restrict__ pwh,
                            const float* __restrict__ bias,
                            float* __restrict__ out) {
    const int lane  = threadIdx.x & 31;
    const int flat  = blockIdx.x * 8 + (threadIdx.x >> 5);
    const int coutT = (flat & 7) << 4;
    const int pixT  = (flat >> 3) << 4;
    v8f c = {};
    c = wmma_kloop(d2 + (size_t)pixT * COUTc, COUTc,
                   pwh + (size_t)coutT * COUTc, COUTc, lane, c);
    const int colc = coutT + (lane & 15);
    const float bb = bias[colc];
    const int prow = pixT + ((lane >> 4) << 3);
#pragma unroll
    for (int r = 0; r < 8; ++r) {
        const int pix = prow + r;
        const int b = pix >> 14, hw = pix & (HWc - 1);
        out[(size_t)(b * COUTc + colc) * HWc + hw] = c[r] + bb;
    }
}

// ---------------------------------------------------------------------------
extern "C" void kernel_launch(void* const* d_in, const int* in_sizes, int n_in,
                              void* d_out, int out_size, void* d_ws, size_t ws_size,
                              hipStream_t stream) {
    const float* x     = (const float*)d_in[0];
    const float* grads = (const float*)d_in[1];
    const float* w1    = (const float*)d_in[2];
    const float* w2    = (const float*)d_in[3];
    const float* mdw   = (const float*)d_in[4];
    const float* mpw   = (const float*)d_in[5];
    const float* mb    = (const float*)d_in[6];
    const float* pdw   = (const float*)d_in[7];
    const float* ppw   = (const float*)d_in[8];
    const float* pb    = (const float*)d_in[9];
    float* out = (float*)d_out;

    char* ws = (char*)d_ws;
    size_t off = 0;
    auto alloc = [&](size_t bytes) {
        char* p = ws + off;
        off = (off + bytes + 255) & ~(size_t)255;
        return p;
    };
    _Float16* XT   = (_Float16*)alloc((size_t)NPIX * CINc * 2);   // reused as DCAT
    _Float16* W1H  = (_Float16*)alloc((size_t)COUTc * CINc * 2);
    _Float16* W2H  = (_Float16*)alloc((size_t)COUTc * CINc * 2);
    _Float16* MPWH = (_Float16*)alloc((size_t)COUTc * 256 * 2);
    _Float16* PPWH = (_Float16*)alloc((size_t)COUTc * COUTc * 2);
    float* M1  = (float*)alloc((size_t)NPIX * COUTc * 4);         // ping (stream1)
    float* M2  = (float*)alloc((size_t)NPIX * COUTc * 4);         // ping (stream2)
    float* P1B = (float*)alloc((size_t)NPIX * COUTc * 4);         // pong / reused as T
    float* P2B = (float*)alloc((size_t)NPIX * COUTc * 4);         // pong / reused as D2
    float* S1  = (float*)alloc((size_t)NPIX * COUTc * 4);
    float* S2  = (float*)alloc((size_t)NPIX * COUTc * 4);
    int4*   IDX1 = (int4*)  alloc((size_t)NPIX * 16);
    float4* WT1  = (float4*)alloc((size_t)NPIX * 16);
    int4*   IDX2 = (int4*)  alloc((size_t)NPIX * 16);
    float4* WT2  = (float4*)alloc((size_t)NPIX * 16);

    // 1) f16 weight packs
    k_f32_to_f16<<<(COUTc * CINc + 255) / 256, 256, 0, stream>>>(w1, W1H, COUTc * CINc);
    k_f32_to_f16<<<(COUTc * CINc + 255) / 256, 256, 0, stream>>>(w2, W2H, COUTc * CINc);
    k_f32_to_f16<<<(COUTc * 256 + 255) / 256, 256, 0, stream>>>(mpw, MPWH, COUTc * 256);
    k_f32_to_f16<<<(COUTc * COUTc + 255) / 256, 256, 0, stream>>>(ppw, PPWH, COUTc * COUTc);

    // 2) NCHW f32 -> NHWC f16
    k_transpose_x<<<dim3(HWc / 32, CINc / 32, BN), dim3(32, 8), 0, stream>>>(x, XT);

    // 3) WMMA 1x1 convs + sigmoid  (2048 tiles x 8 cout-tiles, 8 waves/block)
    k_conv1x1_sigmoid<<<2048, 256, 0, stream>>>(XT, W1H, M1);
    k_conv1x1_sigmoid<<<2048, 256, 0, stream>>>(XT, W2H, M2);

    // 4) constant warp tables
    k_grid_setup<<<NPIX / 256, 256, 0, stream>>>(grads, IDX1, WT1, IDX2, WT2);

    // 5) 24-step warp scan (ping-pong; grid-wide dep -> one launch per step)
    float* pa1 = M1; float* pb1 = P1B; float* pa2 = M2; float* pb2 = P2B;
    for (int t = 0; t < NITER; ++t) {
        k_warp_step<<<(NPIX * 2) / 8, 256, 0, stream>>>(
            (const float4*)pa1, (float4*)pb1, (float4*)S1,
            (const float4*)pa2, (float4*)pb2, (float4*)S2,
            IDX1, WT1, IDX2, WT2, t == 0 ? 1 : 0);
        float* tmp;
        tmp = pa1; pa1 = pb1; pb1 = tmp;
        tmp = pa2; pa2 = pb2; pb2 = tmp;
    }

    // 6) sigmoid + concat + depthwise3x3 -> DCAT (reuse XT region, f16)
    k_dw_merge<<<(NPIX * 256) / 256, 256, 0, stream>>>(S1, S2, mdw, XT);

    // 7) merge pointwise (WMMA) + bias -> T f16 (reuse P1B region)
    k_gemm_merge<<<2048, 256, 0, stream>>>(XT, MPWH, mb, (_Float16*)P1B);

    // 8) post depthwise3x3 -> D2 f16 (reuse P2B region)
    k_dw_post<<<(NPIX * 128) / 256, 256, 0, stream>>>((const _Float16*)P1B, pdw,
                                                      (_Float16*)P2B);

    // 9) post pointwise (WMMA) + bias -> NCHW f32 d_out
    k_gemm_post<<<2048, 256, 0, stream>>>((const _Float16*)P2B, PPWH, pb, out);
}
